// SDREngramMemory_48146583388742
// MI455X (gfx1250) — matrix operand order repaired and checked by reading, not compiled, not run on hardware
//
#include <hip/hip_runtime.h>
#include <hip/hip_bf16.h>

// ---------------- problem constants (match reference) ----------------
constexpr int   BQ      = 1024;    // queries
constexpr int   KDIM    = 2048;    // SDR size
constexpr int   CAP     = 100000;  // bank capacity
constexpr int   CDIM    = 384;     // content dim
constexpr int   TOPK    = 5;
constexpr float INV_NA  = 1.0f / 40.0f;
constexpr float THRESH  = 0.3f;

// ---------------- tiling ----------------
constexpr int MT      = 16;              // queries per block
constexpr int NGRP    = 256;             // bank rows per group (8 waves x 32)
constexpr int GROUPS  = 8;               // groups per block
constexpr int RPB     = NGRP * GROUPS;   // 2048 bank rows per block
constexpr int NCH     = (CAP + RPB - 1) / RPB;  // 49 chunks
constexpr int KT      = 32;              // K per WMMA step
constexpr int KSTEPS  = KDIM / KT;       // 64
constexpr int APAD    = 2056;            // A row stride in bf16 (16B aligned)

typedef __attribute__((ext_vector_type(16))) __bf16 v16bf;
typedef __attribute__((ext_vector_type(8)))  __bf16 v8bf;
typedef __attribute__((ext_vector_type(8)))  float  v8f;
typedef __attribute__((ext_vector_type(4)))  float  v4f;

// =====================================================================
// fp32 -> bf16 bulk convert (grid-stride, 8 elements per iteration)
// =====================================================================
__global__ __launch_bounds__(256)
void cvt_f32_bf16(const float* __restrict__ src, __bf16* __restrict__ dst, long n8)
{
    long i      = (long)blockIdx.x * 256 + threadIdx.x;
    long stride = (long)gridDim.x * 256;
    for (; i < n8; i += stride) {
        const float* s = src + i * 8;
        v4f x = *(const v4f*)(s);
        v4f y = *(const v4f*)(s + 4);
        v8bf o;
#pragma unroll
        for (int j = 0; j < 4; ++j) {
            o[j]     = (__bf16)x[j];
            o[4 + j] = (__bf16)y[j];
        }
        *(v8bf*)(dst + i * 8) = o;
    }
}

// =====================================================================
// Kernel 1: bf16-WMMA GEMM + fused on-the-fly top-5.
//   PRECONV=true : B streamed as bf16 from workspace (2 loads/frag, 0 cvt)
//   PRECONV=false: B streamed as fp32 from input (4 loads + 8 cvt/frag)
// A tile staged once in LDS as bf16. Two accumulators (32 cols) per wave.
// No barriers in the K loop.
// =====================================================================
template <bool PRECONV>
__global__ __launch_bounds__(256)
void sdr_gemm_topk(const float*  __restrict__ qf,
                   const __bf16* __restrict__ qb,
                   const float*  __restrict__ bankf,
                   const __bf16* __restrict__ bankb,
                   const unsigned char* __restrict__ valid,
                   float* __restrict__ cand_val,
                   int*   __restrict__ cand_idx)
{
    __shared__ __align__(16) __bf16 Ald[MT][APAD];   // 64 KB bf16 query tile
    __shared__ __align__(16) float  sims[MT][260];   // f32 sims, padded
    __shared__ float cvals[MT][8][TOPK];
    __shared__ int   cidxs[MT][8][TOPK];

    const int tid  = threadIdx.x;
    const int lane = tid & 31;
    const int wav  = tid >> 5;
    const int half = lane >> 4;
    const int lm   = lane & 15;
    const int  qbase     = blockIdx.x * MT;
    const long chunkbase = (long)blockIdx.y * RPB;

    // ---- one-time A stage into LDS (16 x 2048 bf16) ----
    {
        const int row = tid >> 4;            // 0..15
        const int kb  = (tid & 15) << 7;     // 0,128,...,1920
        if constexpr (PRECONV) {
            const __bf16* src = qb + (long)(qbase + row) * KDIM + kb;
#pragma unroll
            for (int i = 0; i < 128; i += 8)
                *(v8bf*)&Ald[row][kb + i] = *(const v8bf*)(src + i);
        } else {
            const float* src = qf + (long)(qbase + row) * KDIM + kb;
#pragma unroll
            for (int i = 0; i < 128; i += 8) {
                v4f x = *(const v4f*)(src + i);
                v4f y = *(const v4f*)(src + i + 4);
                v8bf o;
#pragma unroll
                for (int j = 0; j < 4; ++j) {
                    o[j]     = (__bf16)x[j];
                    o[4 + j] = (__bf16)y[j];
                }
                *(v8bf*)&Ald[row][kb + i] = o;
            }
        }
    }
    __syncthreads();

    // running per-thread top-5 (threads 0..127: 8 scan threads per query)
    float tv[TOPK]; int ti[TOPK];
#pragma unroll
    for (int j = 0; j < TOPK; ++j) { tv[j] = -__builtin_inff(); ti[j] = 0x7fffffff; }

    for (int g = 0; g < GROUPS; ++g) {
        const long rowbase = chunkbase + (long)g * NGRP;

        // two bank rows per lane (clamped; OOB excluded in scan)
        long r0 = rowbase + wav * 32 + lm;
        long r1 = r0 + 16;
        if (r0 > (long)(CAP - 1)) r0 = CAP - 1;
        if (r1 > (long)(CAP - 1)) r1 = CAP - 1;

        const __bf16* bb0 = PRECONV ? (bankb + r0 * (long)KDIM + half * 16) : nullptr;
        const __bf16* bb1 = PRECONV ? (bankb + r1 * (long)KDIM + half * 16) : nullptr;
        const float*  bf0 = PRECONV ? nullptr : (bankf + r0 * (long)KDIM + half * 16);
        const float*  bf1 = PRECONV ? nullptr : (bankf + r1 * (long)KDIM + half * 16);

        v8f c0 = {};
        v8f c1 = {};
#pragma unroll 2
        for (int t = 0; t < KSTEPS; ++t) {
            // A fragment from LDS (contiguous k = 16*half + s, matching B)
            v8bf alo = *(const v8bf*)&Ald[lm][t * KT + half * 16];
            v8bf ahi = *(const v8bf*)&Ald[lm][t * KT + half * 16 + 8];
            v16bf a = __builtin_shufflevector(alo, ahi,
                0,1,2,3,4,5,6,7,8,9,10,11,12,13,14,15);

            v16bf b0, b1;
            if constexpr (PRECONV) {
                v8bf l0 = *(const v8bf*)(bb0 + t * KT);
                v8bf h0 = *(const v8bf*)(bb0 + t * KT + 8);
                v8bf l1 = *(const v8bf*)(bb1 + t * KT);
                v8bf h1 = *(const v8bf*)(bb1 + t * KT + 8);
                b0 = __builtin_shufflevector(l0, h0,
                    0,1,2,3,4,5,6,7,8,9,10,11,12,13,14,15);
                b1 = __builtin_shufflevector(l1, h1,
                    0,1,2,3,4,5,6,7,8,9,10,11,12,13,14,15);
            } else {
                const float* p0 = bf0 + t * KT;
                const float* p1 = bf1 + t * KT;
                v4f f0 = *(const v4f*)(p0);     v4f f1 = *(const v4f*)(p0 + 4);
                v4f f2 = *(const v4f*)(p0 + 8); v4f f3 = *(const v4f*)(p0 + 12);
                v4f g0 = *(const v4f*)(p1);     v4f g1 = *(const v4f*)(p1 + 4);
                v4f g2 = *(const v4f*)(p1 + 8); v4f g3 = *(const v4f*)(p1 + 12);
#pragma unroll
                for (int i = 0; i < 4; ++i) {
                    b0[i]      = (__bf16)f0[i];
                    b0[4 + i]  = (__bf16)f1[i];
                    b0[8 + i]  = (__bf16)f2[i];
                    b0[12 + i] = (__bf16)f3[i];
                    b1[i]      = (__bf16)g0[i];
                    b1[4 + i]  = (__bf16)g1[i];
                    b1[8 + i]  = (__bf16)g2[i];
                    b1[12 + i] = (__bf16)g3[i];
                }
            }
            c0 = __builtin_amdgcn_wmma_f32_16x16x32_bf16(
                     false, a, false, b0, (short)0, c0, false, false);
            c1 = __builtin_amdgcn_wmma_f32_16x16x32_bf16(
                     false, a, false, b1, (short)0, c1, false, false);
        }

        __syncthreads();   // previous scan finished reading sims
        // dump C (lanes 0-15: M=v; lanes 16-31: M=v+8; N = lane%16)
#pragma unroll
        for (int v = 0; v < 8; ++v) {
            int m = (lane < 16) ? v : (v + 8);
            sims[m][wav * 32 + lm]      = c0[v];
            sims[m][wav * 32 + 16 + lm] = c1[v];
        }
        __syncthreads();
        // scan: 8 threads per query, 32 cols each, maintain running top-5
        if (tid < 128) {
            const int m  = tid >> 3;
            const int cb = (tid & 7) * 32;
            for (int cc = 0; cc < 32; ++cc) {
                const int  col  = cb + cc;
                const long gcol = rowbase + col;
                if (gcol >= CAP) continue;
                if (!valid[gcol]) continue;
                float v = sims[m][col] * INV_NA;
                if (v > tv[TOPK - 1]) {
                    tv[TOPK - 1] = v; ti[TOPK - 1] = (int)gcol;
#pragma unroll
                    for (int j = TOPK - 1; j > 0; --j) {
                        if (tv[j] > tv[j - 1]) {
                            float fv = tv[j]; tv[j] = tv[j - 1]; tv[j - 1] = fv;
                            int   ii = ti[j]; ti[j] = ti[j - 1]; ti[j - 1] = ii;
                        }
                    }
                }
            }
        }
    }

    // block epilogue: merge 8x5 -> exact top-5 per query, write workspace
    if (tid < 128) {
        const int m = tid >> 3, seg = tid & 7;
#pragma unroll
        for (int j = 0; j < TOPK; ++j) { cvals[m][seg][j] = tv[j]; cidxs[m][seg][j] = ti[j]; }
    }
    __syncthreads();
    if (tid < MT) {
        float mv[TOPK]; int mi[TOPK];
#pragma unroll
        for (int j = 0; j < TOPK; ++j) { mv[j] = -__builtin_inff(); mi[j] = 0x7fffffff; }
        for (int s = 0; s < 8; ++s) {
            for (int j = 0; j < TOPK; ++j) {
                float v  = cvals[tid][s][j];
                int   ix = cidxs[tid][s][j];
                bool ins = (v > mv[TOPK - 1]) ||
                           (v == mv[TOPK - 1] && ix < mi[TOPK - 1]);
                if (ins) {
                    mv[TOPK - 1] = v; mi[TOPK - 1] = ix;
#pragma unroll
                    for (int k = TOPK - 1; k > 0; --k) {
                        bool up = (mv[k] > mv[k - 1]) ||
                                  (mv[k] == mv[k - 1] && mi[k] < mi[k - 1]);
                        if (up) {
                            float fv = mv[k]; mv[k] = mv[k - 1]; mv[k - 1] = fv;
                            int   ii = mi[k]; mi[k] = mi[k - 1]; mi[k - 1] = ii;
                        }
                    }
                }
            }
        }
        const long base = ((long)(qbase + tid) * NCH + blockIdx.y) * TOPK;
#pragma unroll
        for (int j = 0; j < TOPK; ++j) { cand_val[base + j] = mv[j]; cand_idx[base + j] = mi[j]; }
    }
}

// =====================================================================
// Kernel 2: reduce 49*5 candidates per query -> top-5, threshold, gather
// =====================================================================
__global__ __launch_bounds__(256)
void sdr_finalize(const float* __restrict__ cand_val,
                  const int*   __restrict__ cand_idx,
                  const float* __restrict__ content,
                  float* __restrict__ out)
{
    constexpr int NC = NCH * TOPK;  // 245 candidates
    const int qi  = blockIdx.x;
    const int tid = threadIdx.x;

    __shared__ float vals[256]; __shared__ int idxs[256];
    __shared__ float rv[256];   __shared__ int rb[256]; __shared__ int rp[256];
    __shared__ float selv[TOPK]; __shared__ int seli[TOPK];

    const long cbase = (long)qi * NC;
    if (tid < NC) { vals[tid] = cand_val[cbase + tid]; idxs[tid] = cand_idx[cbase + tid]; }
    else          { vals[tid] = -__builtin_inff();     idxs[tid] = 0x7fffffff; }
    __syncthreads();

    for (int j = 0; j < TOPK; ++j) {
        rv[tid] = vals[tid]; rb[tid] = idxs[tid]; rp[tid] = tid;
        __syncthreads();
        for (int s = 128; s > 0; s >>= 1) {
            if (tid < s) {
                float va = rv[tid], vb = rv[tid + s];
                int   ba = rb[tid], bb = rb[tid + s];
                if (vb > va || (vb == va && bb < ba)) {
                    rv[tid] = vb; rb[tid] = bb; rp[tid] = rp[tid + s];
                }
            }
            __syncthreads();
        }
        if (tid == 0) {
            selv[j] = rv[0]; seli[j] = rb[0];
            vals[rp[0]] = -__builtin_inff(); idxs[rp[0]] = 0x7fffffff;
        }
        __syncthreads();
    }

    const long SIM_OFF = (long)BQ * TOPK * CDIM;
    if (tid < TOPK) {
        float v = selv[tid];
        out[SIM_OFF + (long)qi * TOPK + tid] = (v >= THRESH) ? v : 0.f;
    }
#pragma unroll
    for (int j = 0; j < TOPK; ++j) {
        const bool keep = selv[j] >= THRESH;
        int src = seli[j];
        if (src < 0 || src >= CAP) src = 0;
        const float* cp = content + (long)src * CDIM;
        float* op = out + ((long)qi * TOPK + j) * CDIM;
        for (int e = tid; e < CDIM; e += 256)
            op[e] = keep ? cp[e] : 0.f;
    }
}

// =====================================================================
extern "C" void kernel_launch(void* const* d_in, const int* in_sizes, int n_in,
                              void* d_out, int out_size, void* d_ws, size_t ws_size,
                              hipStream_t stream)
{
    const float*         q       = (const float*)d_in[0];
    const float*         bank    = (const float*)d_in[1];
    const float*         content = (const float*)d_in[2];
    const unsigned char* valid   = (const unsigned char*)d_in[3];
    // d_in[4] = top_k scalar (compile-time TOPK here)
    float* out = (float*)d_out;

    // workspace layout: [cand_val | cand_idx | q_bf16 | bank_bf16]
    const size_t n_cand     = (size_t)BQ * NCH * TOPK;          // 250,880
    const size_t cand_bytes = n_cand * (sizeof(float) + sizeof(int));
    const size_t qbf_off    = (cand_bytes + 255) & ~(size_t)255;
    const size_t qbf_bytes  = (size_t)BQ * KDIM * sizeof(__bf16);
    const size_t bank_off   = (qbf_off + qbf_bytes + 255) & ~(size_t)255;
    const size_t bank_bytes = (size_t)CAP * KDIM * sizeof(__bf16);
    const size_t need       = bank_off + bank_bytes;

    float*  cand_val = (float*)d_ws;
    int*    cand_idx = (int*)((char*)d_ws + n_cand * sizeof(float));
    __bf16* qbf      = (__bf16*)((char*)d_ws + qbf_off);
    __bf16* bankbf   = (__bf16*)((char*)d_ws + bank_off);

    dim3 g1(BQ / MT, NCH);   // chunk-major ordering keeps bank panel L2-hot

    if (ws_size >= need) {
        // pre-convert bank + queries to bf16 (memory-bound, ~1.2 GB total)
        cvt_f32_bf16<<<4096, 256, 0, stream>>>(bank, bankbf, (long)CAP * KDIM / 8);
        cvt_f32_bf16<<<512,  256, 0, stream>>>(q,    qbf,    (long)BQ  * KDIM / 8);
        sdr_gemm_topk<true><<<g1, 256, 0, stream>>>(
            q, qbf, bank, bankbf, valid, cand_val, cand_idx);
    } else {
        sdr_gemm_topk<false><<<g1, 256, 0, stream>>>(
            q, nullptr, bank, nullptr, valid, cand_val, cand_idx);
    }
    sdr_finalize<<<BQ, 256, 0, stream>>>(cand_val, cand_idx, content, out);
}
